// EdgeNodeFusion_33646773797597
// MI455X (gfx1250) — matrix operand order repaired and checked by reading, not compiled, not run on hardware
//
#include <hip/hip_runtime.h>
#include <hip/hip_bf16.h>

// ---------------------------------------------------------------------------
// Problem constants (from reference)
// ---------------------------------------------------------------------------
#define NNODE   2048
#define E_RAW   2200000
#define E_MASK  2096128        // N*(N-1)/2, == 16 * 131008
#define E_ADJ   65536
#define F_IN    32
#define D_EDGE  32
#define F_NODE  128
#define F_OUT   64

#define SALT_KEY1 0x9E3779B9u
#define SALT_KEY2 0x3C6EF372u
#define KEEP16    58982u       // floor(0.9 * 65536); p_keep = 0.89999
#define KEEP_SCL  (1.0f / 0.9f)

typedef __attribute__((ext_vector_type(16))) _Float16 v16h;
typedef __attribute__((ext_vector_type(8)))  float    v8f;

// ---------------------------------------------------------------------------
// gfx1250 async global->LDS staging (ASYNCcnt path).
// Probe (round 2 stderr) showed the builtin takes pointers to a 4-int vector:
//   param0: v4i in AS1 (global), param1: v4i in AS3 (LDS), then imm offset/cpol.
// ---------------------------------------------------------------------------
#if __has_builtin(__builtin_amdgcn_global_load_async_to_lds_b128) && \
    __has_builtin(__builtin_amdgcn_s_wait_asynccnt)
#define ASYNC_LDS_OK 1
typedef __attribute__((vector_size(4 * sizeof(int)))) int v4i_vs;
typedef __attribute__((address_space(1))) v4i_vs* gv4i_p;
typedef __attribute__((address_space(3))) v4i_vs* lv4i_p;
#endif

__device__ __forceinline__ void stage_f32_to_lds(const float* __restrict__ g,
                                                 float* __restrict__ l,
                                                 int n, int tid, int nthreads) {
#ifdef ASYNC_LDS_OK
    for (int i = tid * 4; i < n; i += nthreads * 4) {
        __builtin_amdgcn_global_load_async_to_lds_b128(
            (gv4i_p)(g + i), (lv4i_p)(l + i), 0, 0);
    }
    __builtin_amdgcn_s_wait_asynccnt(0);
#else
    for (int i = tid; i < n; i += nthreads) l[i] = g[i];
#endif
}

// ---------------------------------------------------------------------------
// Cheap counter-based PRNG for dropout (murmur3 finalizer, ~6 VALU ops).
// Keyed on the PRE-MASK element index so duplicated gather rows get identical
// masks, consistently across kernels. One 32-bit hash yields two independent
// 16-bit Bernoulli draws (for columns c and c+16 / c+32).
// ---------------------------------------------------------------------------
__device__ __forceinline__ unsigned mix32(unsigned x) {
    x ^= x >> 16; x *= 0x85EBCA6Bu;
    x ^= x >> 13; x *= 0xC2B2AE35u;
    x ^= x >> 16;
    return x;
}

__device__ __forceinline__ float lrelu(float v) {
    return v >= 0.0f ? v : 0.01f * v;
}

__device__ __forceinline__ v16h pack16(float4 q0, float4 q1, float4 q2, float4 q3) {
    v16h a;
    a[0]  = (_Float16)q0.x; a[1]  = (_Float16)q0.y; a[2]  = (_Float16)q0.z; a[3]  = (_Float16)q0.w;
    a[4]  = (_Float16)q1.x; a[5]  = (_Float16)q1.y; a[6]  = (_Float16)q1.z; a[7]  = (_Float16)q1.w;
    a[8]  = (_Float16)q2.x; a[9]  = (_Float16)q2.y; a[10] = (_Float16)q2.z; a[11] = (_Float16)q2.w;
    a[12] = (_Float16)q3.x; a[13] = (_Float16)q3.y; a[14] = (_Float16)q3.z; a[15] = (_Float16)q3.w;
    return a;
}

// ---------------------------------------------------------------------------
// Kernel: zero a float region (harness poisons ws; atomics need zeros)
// ---------------------------------------------------------------------------
__global__ void zero_kernel(float* __restrict__ p, int n) {
    int i = blockIdx.x * blockDim.x + threadIdx.x;
    if (i < n) p[i] = 0.0f;
}

// ---------------------------------------------------------------------------
// Kernel 1 (dominant, HBM-bound: ~0.55 GB -> ~24 us at 23.3 TB/s):
//   out_e[r,:] = (1 + b_ln) * dropout(leaky(edge_attr[edge_mask[r]] @ W_dg + b_dg))
// LayerNorm over a size-1 axis is the constant b_ln, so the whole NxN gram /
// sigmoid / triu path is dead code -> out_e = (1+b_ln)*e. e is recomputed from
// the gathered edge_attr rows (same 128 B/row traffic as re-reading a
// materialized e, so materializing e would only add 2x281 MB of traffic).
// One wave = one 16-row tile via two v_wmma_f32_16x16x32_f16 (K=32 exactly).
// ---------------------------------------------------------------------------
__global__ __launch_bounds__(256) void edge_out_kernel(
    const float* __restrict__ edge_attr,   // [E_RAW, 32]
    const int*   __restrict__ edge_mask,   // [E_MASK]
    const float* __restrict__ W_dg,        // [32, 32]
    const float* __restrict__ b_dg,        // [32]
    const float* __restrict__ b_ln,        // [1]
    float*       __restrict__ out_e)       // [E_MASK, 32]
{
    __shared__ __align__(16) float sW[32 * 32];   // 4 KB, async-staged
    __shared__ int sMask[8][16];

    int tid = threadIdx.x;
    stage_f32_to_lds(W_dg, sW, 32 * 32, tid, 256);

    int warp = tid >> 5, lane = tid & 31;
    int tile = blockIdx.x * 8 + warp;        // 16376 * 8 == 131008 tiles exactly
    int r0   = tile * 16;
    if (lane < 16) sMask[warp][lane] = edge_mask[r0 + lane];
    __syncthreads();

    int m = lane & 15, half = lane >> 4;

    // A: 16x32 f16; lane(m,half) holds K in [half*16,+16)
    int row = sMask[warp][m];
    const float4* pa = (const float4*)(edge_attr + (size_t)row * 32 + half * 16);
    float4 q0 = pa[0], q1 = pa[1];
    const float4* pb = (const float4*)(edge_attr + (size_t)row * 32 + half * 16 + 8);
    float4 q2 = pb[0], q3 = pb[1];
    v16h a = pack16(q0, q1, q2, q3);

    // B: 32x16 f16 per WMMA; lane(n=m, khalf=half): b[i] = W[half*16+i][col]
    v16h b0, b1;
#pragma unroll
    for (int i = 0; i < 16; ++i) {
        b0[i] = (_Float16)sW[(half * 16 + i) * 32 + m];
        b1[i] = (_Float16)sW[(half * 16 + i) * 32 + 16 + m];
    }

    float bias0 = b_dg[m], bias1 = b_dg[16 + m];
    v8f c0, c1;
#pragma unroll
    for (int r = 0; r < 8; ++r) { c0[r] = bias0; c1[r] = bias1; }

    c0 = __builtin_amdgcn_wmma_f32_16x16x32_f16(false, a, false, b0, (short)0, c0, false, false);
    c1 = __builtin_amdgcn_wmma_f32_16x16x32_f16(false, a, false, b1, (short)0, c1, false, false);

    float scale = 1.0f + b_ln[0];

    // D layout: lane -> N = m, M = r + 8*half. One hash per (lane,row) feeds
    // both columns (n and n+16) via 16-bit slices: ~5 VALU ops per element.
#pragma unroll
    for (int r = 0; r < 8; ++r) {
        int rr = r + 8 * half;
        int eRow = sMask[warp][rr];                 // pre-mask row: dropout key
        size_t obase = (size_t)(r0 + rr) * 32;
        unsigned hsh = mix32(((unsigned)eRow * 16u + (unsigned)m) ^ SALT_KEY1);
        float v0 = lrelu(c0[r]);
        v0 = ((hsh & 0xFFFFu) < KEEP16) ? v0 * KEEP_SCL : 0.0f;
        float v1 = lrelu(c1[r]);
        v1 = ((hsh >> 16) < KEEP16) ? v1 * KEEP_SCL : 0.0f;
        out_e[obase + m]      = scale * v0;
        out_e[obase + 16 + m] = scale * v1;
    }
}

// ---------------------------------------------------------------------------
// Order-preserving compaction of adjacency_indices -> sel[E_ADJ]
// ---------------------------------------------------------------------------
__global__ __launch_bounds__(256) void count_kernel(const int* __restrict__ adj,
                                                    int total,
                                                    unsigned* __restrict__ counts) {
    __shared__ unsigned sc[256];
    int t = threadIdx.x;
    long base = (long)blockIdx.x * 8192 + (long)t * 32;
    unsigned c = 0;
    for (int k = 0; k < 32; ++k) {
        long i = base + k;
        if (i < total) c += (adj[i] != 0) ? 1u : 0u;
    }
    sc[t] = c;
    __syncthreads();
    for (int s = 128; s > 0; s >>= 1) {
        if (t < s) sc[t] += sc[t + s];
        __syncthreads();
    }
    if (t == 0) counts[blockIdx.x] = sc[0];
}

__global__ void scan_kernel(const unsigned* __restrict__ counts,
                            unsigned* __restrict__ offs, int nblk) {
    if (threadIdx.x == 0 && blockIdx.x == 0) {
        unsigned run = 0;
        for (int i = 0; i < nblk; ++i) { offs[i] = run; run += counts[i]; }
    }
}

__global__ __launch_bounds__(256) void scatter_kernel(const int* __restrict__ adj,
                                                      int total,
                                                      const unsigned* __restrict__ offs,
                                                      int* __restrict__ sel) {
    __shared__ unsigned sc[256];
    int t = threadIdx.x;
    long base = (long)blockIdx.x * 8192 + (long)t * 32;
    unsigned c = 0;
    for (int k = 0; k < 32; ++k) {
        long i = base + k;
        if (i < total) c += (adj[i] != 0) ? 1u : 0u;
    }
    sc[t] = c;
    __syncthreads();
    if (t == 0) {
        unsigned run = 0;
        for (int i = 0; i < 256; ++i) { unsigned v = sc[i]; sc[i] = run; run += v; }
    }
    __syncthreads();
    unsigned off = offs[blockIdx.x] + sc[t];
    for (int k = 0; k < 32; ++k) {
        long i = base + k;
        if (i < total && adj[i] != 0) sel[off++] = (int)i;
    }
}

// ---------------------------------------------------------------------------
// w[r] = mean over 32 feats of recomputed e row for selected edges (tiny).
// Uses the SAME pre-mask-index dropout keying as edge_out_kernel.
// ---------------------------------------------------------------------------
__global__ __launch_bounds__(256) void wsel_kernel(
    const float* __restrict__ edge_attr, const int* __restrict__ edge_mask,
    const int* __restrict__ sel, const float* __restrict__ W_dg,
    const float* __restrict__ b_dg, float* __restrict__ w)
{
    __shared__ __align__(16) float sW[32 * 32];
    __shared__ float sB[32];
    int tid = threadIdx.x;
    stage_f32_to_lds(W_dg, sW, 32 * 32, tid, 256);
    if (tid < 32) sB[tid] = b_dg[tid];
    __syncthreads();

    int r = blockIdx.x * 256 + tid;
    if (r >= E_ADJ) return;
    int i   = sel[r];
    int row = edge_mask[i];
    const float* ar = edge_attr + (size_t)row * 32;
    float av[32];
#pragma unroll
    for (int k = 0; k < 32; ++k) av[k] = ar[k];
    float sum = 0.0f;
    for (int f = 0; f < 16; ++f) {
        float acc0 = sB[f], acc1 = sB[f + 16];
#pragma unroll
        for (int k = 0; k < 32; ++k) {
            acc0 += av[k] * sW[k * 32 + f];
            acc1 += av[k] * sW[k * 32 + f + 16];
        }
        unsigned hsh = mix32(((unsigned)row * 16u + (unsigned)f) ^ SALT_KEY1);
        acc0 = lrelu(acc0);
        acc1 = lrelu(acc1);
        sum += ((hsh & 0xFFFFu) < KEEP16) ? acc0 * KEEP_SCL : 0.0f;
        sum += ((hsh >> 16)     < KEEP16) ? acc1 * KEEP_SCL : 0.0f;
    }
    w[r] = sum * (1.0f / 32.0f);
}

// ---------------------------------------------------------------------------
// GCN normalization: degree (both directions; self-loop weight 1 added later)
// ---------------------------------------------------------------------------
__global__ void degree_kernel(const int* __restrict__ ei, const float* __restrict__ w,
                              float* __restrict__ deg) {
    int r = blockIdx.x * blockDim.x + threadIdx.x;
    if (r >= E_ADJ) return;
    int s = ei[r], d = ei[E_ADJ + r];
    float wv = w[r];
    atomicAdd(deg + d, wv);
    atomicAdd(deg + s, wv);
}

__global__ void dinv_kernel(const float* __restrict__ deg, float* __restrict__ dinv) {
    int n = blockIdx.x * blockDim.x + threadIdx.x;
    if (n >= NNODE) return;
    float dv = deg[n] + 1.0f;          // +1 for self loop
    dinv[n] = (dv > 0.0f) ? rsqrtf(dv) : 0.0f;
}

// ---------------------------------------------------------------------------
// h = node_attr @ W_gcn  (2048x128 @ 128x64) via WMMA f16, 4 K-steps.
// W_gcn async-staged to LDS as f32 (32 KB), converted at fragment build.
// ---------------------------------------------------------------------------
__global__ __launch_bounds__(256) void hgemm_kernel(
    const float* __restrict__ node_attr,   // [2048, 128]
    const float* __restrict__ W_gcn,       // [128, 64]
    float*       __restrict__ h)           // [2048, 64]
{
    __shared__ __align__(16) float sW[F_NODE * F_OUT];   // 32 KB
    int tid = threadIdx.x;
    stage_f32_to_lds(W_gcn, sW, F_NODE * F_OUT, tid, 256);
    __syncthreads();

    int warp = tid >> 5, lane = tid & 31;
    int tile = blockIdx.x * 8 + warp;       // 64 blocks * 8 = 512 = 128 x 4 tiles
    int rt = tile >> 2, ct = tile & 3;
    int m = lane & 15, half = lane >> 4;

    v8f c;
#pragma unroll
    for (int r = 0; r < 8; ++r) c[r] = 0.0f;

    const float* arow = node_attr + (size_t)(rt * 16 + m) * F_NODE;
#pragma unroll
    for (int kk = 0; kk < 4; ++kk) {
        const float4* pa = (const float4*)(arow + kk * 32 + half * 16);
        float4 q0 = pa[0], q1 = pa[1];
        const float4* pb = (const float4*)(arow + kk * 32 + half * 16 + 8);
        float4 q2 = pb[0], q3 = pb[1];
        v16h a = pack16(q0, q1, q2, q3);
        v16h b;
#pragma unroll
        for (int i = 0; i < 16; ++i)
            b[i] = (_Float16)sW[(kk * 32 + half * 16 + i) * F_OUT + ct * 16 + m];
        c = __builtin_amdgcn_wmma_f32_16x16x32_f16(false, a, false, b, (short)0, c, false, false);
    }
#pragma unroll
    for (int r = 0; r < 8; ++r)
        h[(size_t)(rt * 16 + r + 8 * half) * F_OUT + ct * 16 + m] = c[r];
}

// ---------------------------------------------------------------------------
// Scatter-aggregate: xacc[d] += norm*h[s] and xacc[s] += norm*h[d]
// ---------------------------------------------------------------------------
__global__ __launch_bounds__(256) void aggregate_kernel(
    const int* __restrict__ ei, const float* __restrict__ w,
    const float* __restrict__ dinv, const float* __restrict__ h,
    float* __restrict__ xacc)
{
    int t = blockIdx.x * 256 + threadIdx.x;
    int j = t >> 6, f = t & 63;
    if (j >= E_ADJ) return;
    int s = ei[j], d = ei[E_ADJ + j];
    float nm = dinv[s] * w[j] * dinv[d];
    atomicAdd(&xacc[(size_t)d * F_OUT + f], nm * h[(size_t)s * F_OUT + f]);
    atomicAdd(&xacc[(size_t)s * F_OUT + f], nm * h[(size_t)d * F_OUT + f]);
}

// ---------------------------------------------------------------------------
// Finalize x: self-loop + bias + leaky + dropout(key 2) -> second output.
// Hash keyed per (n, f mod 32); slice selected by f/32.
// ---------------------------------------------------------------------------
__global__ __launch_bounds__(256) void finalize_x_kernel(
    const float* __restrict__ xacc, const float* __restrict__ h,
    const float* __restrict__ dinv, const float* __restrict__ b_gcn,
    float* __restrict__ xout)
{
    int t = blockIdx.x * 256 + threadIdx.x;
    if (t >= NNODE * F_OUT) return;
    int n = t >> 6, f = t & 63;
    float dv = dinv[n];
    float v = xacc[t] + dv * dv * h[t] + b_gcn[f];
    v = lrelu(v);
    unsigned hsh = mix32(((unsigned)n * 32u + (unsigned)(f & 31)) ^ SALT_KEY2);
    unsigned u16 = (f >= 32) ? (hsh >> 16) : (hsh & 0xFFFFu);
    v = (u16 < KEEP16) ? v * KEEP_SCL : 0.0f;
    xout[t] = v;
}

// ---------------------------------------------------------------------------
// Host launch
// ---------------------------------------------------------------------------
extern "C" void kernel_launch(void* const* d_in, const int* in_sizes, int n_in,
                              void* d_out, int out_size, void* d_ws, size_t ws_size,
                              hipStream_t stream) {
    (void)in_sizes; (void)n_in; (void)out_size; (void)ws_size;

    const float* node_attr  = (const float*)d_in[0];
    const float* edge_attr  = (const float*)d_in[1];
    const int*   edge_index = (const int*)d_in[2];
    const int*   edge_mask  = (const int*)d_in[3];
    const int*   adj        = (const int*)d_in[4];
    const float* W_dg       = (const float*)d_in[5];
    const float* b_dg       = (const float*)d_in[6];
    const float* W_gcn      = (const float*)d_in[7];
    const float* b_gcn      = (const float*)d_in[8];
    // d_in[9] (gamma) multiplies an exact zero in the reference -> unused
    const float* b_ln       = (const float*)d_in[10];

    float* out_e = (float*)d_out;                         // [E_MASK, 32]
    float* x_out = out_e + (size_t)E_MASK * 32;           // [2048, 64]

    // Workspace layout (~1.6 MB; xacc and deg contiguous so one zero pass works)
    char* p = (char*)d_ws;
    float* xacc = (float*)p;  p += (size_t)NNODE * F_OUT * 4;
    float* deg  = (float*)p;  p += (size_t)NNODE * 4;
    float* dinv = (float*)p;  p += (size_t)NNODE * 4;
    float* h    = (float*)p;  p += (size_t)NNODE * F_OUT * 4;
    float* w    = (float*)p;  p += (size_t)E_ADJ * 4;
    int*   sel  = (int*)p;    p += (size_t)E_ADJ * 4;
    unsigned* bc = (unsigned*)p; p += 256 * 4;
    unsigned* bo = (unsigned*)p; p += 256 * 4;

    int zeroN = NNODE * F_OUT + NNODE;  // xacc + deg
    zero_kernel<<<(zeroN + 255) / 256, 256, 0, stream>>>(xacc, zeroN);

    // Dominant kernel: 131008 tiles of 16 rows, 8 waves/block -> 16376 blocks
    edge_out_kernel<<<E_MASK / 16 / 8, 256, 0, stream>>>(
        edge_attr, edge_mask, W_dg, b_dg, b_ln, out_e);

    count_kernel<<<256, 256, 0, stream>>>(adj, E_MASK, bc);
    scan_kernel<<<1, 32, 0, stream>>>(bc, bo, 256);
    scatter_kernel<<<256, 256, 0, stream>>>(adj, E_MASK, bo, sel);

    wsel_kernel<<<E_ADJ / 256, 256, 0, stream>>>(edge_attr, edge_mask, sel, W_dg, b_dg, w);
    degree_kernel<<<E_ADJ / 256, 256, 0, stream>>>(edge_index, w, deg);
    dinv_kernel<<<NNODE / 256, 256, 0, stream>>>(deg, dinv);

    hgemm_kernel<<<64, 256, 0, stream>>>(node_attr, W_gcn, h);
    aggregate_kernel<<<(E_ADJ * F_OUT) / 256, 256, 0, stream>>>(edge_index, w, dinv, h, xacc);
    finalize_x_kernel<<<(NNODE * F_OUT) / 256, 256, 0, stream>>>(xacc, h, dinv, b_gcn, x_out);
}
